// AILayerNorm_19765439496657
// MI455X (gfx1250) — compile-verified, matrix operand order checked by reading, not run on hardware
//
#include <hip/hip_runtime.h>
#include <stdint.h>

// ---------------------------------------------------------------------------
// AILayerNorm (quantized layernorm), MI455X / gfx1250, wave32.
// Bandwidth-bound: 4 reads of x (L2-resident, 128MB < 192MB L2) + 1 write.
// Pass structure: k_init -> k_absmax(x) -> k_rowstats (TDM-staged tiles,
// WMMA row sums, max|y|) -> k_output (quantize with scale_out).
// ---------------------------------------------------------------------------

typedef __attribute__((ext_vector_type(16))) _Float16     v16h;
typedef __attribute__((ext_vector_type(8)))  float        v8f;
typedef __attribute__((ext_vector_type(4)))  unsigned int u32x4;
typedef __attribute__((ext_vector_type(8)))  int          i32x8;
typedef __attribute__((ext_vector_type(4)))  int          i32x4;

#define N_COLS        4096
#define N_ROWS        8192
#define CHUNK         1024   // columns staged per TDM round (16x1024 f32 = 64 KB LDS)
#define ROWS_PER_BLK  16

__device__ __constant__ float kInvSqrtLut[16] = {
    65535.f, 46341.f, 32768.f, 23170.f, 16384.f, 11585.f, 8192.f, 5793.f,
    4096.f, 2896.f, 2048.f, 1448.f, 1024.f, 724.f, 512.f, 362.f};

// round-to-nearest-even + clip to int8 range (matches jnp.round/clip)
__device__ __forceinline__ float quantInt(float x, float invScale) {
    float r = rintf(x * invScale);
    return fminf(fmaxf(r, -127.f), 127.f);
}

// LUT-decomposed square: sq_ref = (hi ? idx_h^2*16 : idx_m^2) * 2^(2*ALPHA)
__device__ __forceinline__ float sqRefOf(float absq) {
    bool hi = absq >= 64.f;                          // top2 >= 1
    int idxh = min((int)floorf(absq * 0.0625f), 15); // floor(a/16) clip 15
    int idxm = ((int)floorf(absq * 0.5f)) & 15;      // mod(floor(a/2),16)
    int idx  = hi ? idxh : idxm;
    float sq = (float)(idx * idx);
    sq = hi ? sq * 16.f : sq;
    return sq * 16.f;                                // * 2^(2*ALPHA), ALPHA=2
}

__device__ __forceinline__ float waveMax(float v) {
#pragma unroll
    for (int off = 16; off > 0; off >>= 1)
        v = fmaxf(v, __shfl_xor(v, off, 32));        // wave32
    return v;
}

// ---- Tensor Data Mover: 2D tile (tileRows x tileCols f32) global -> LDS ----
__device__ __forceinline__ void tdmLoadTile(unsigned ldsAddr, const float* gsrc,
                                            int tileCols, int tileRows,
                                            long long rowStrideElems) {
    unsigned long long ga = (unsigned long long)(uintptr_t)gsrc;
    u32x4 g0;
    g0[0] = 1u;                                         // count=1, is_restore=0
    g0[1] = ldsAddr;                                    // LDS byte address
    g0[2] = (unsigned)(ga & 0xFFFFFFFFull);             // global_addr[31:0]
    g0[3] = (unsigned)((ga >> 32) & 0x1FFFFFFull)       // global_addr[56:32]
          | (2u << 30);                                 // type=2 ("image")
    unsigned td0 = (unsigned)tileCols;                  // tensor_dim0 (OOB bound)
    unsigned td1 = (unsigned)tileRows;                  // tensor_dim1
    unsigned long long st0 = (unsigned long long)rowStrideElems;
    i32x8 g1;
    g1[0] = (int)(2u << 16);                            // data_size=2 -> 4 bytes
    g1[1] = (int)((td0 & 0xFFFFu) << 16);               // dim0[15:0] (abar addr=0)
    g1[2] = (int)((td0 >> 16) | ((td1 & 0xFFFFu) << 16));
    g1[3] = (int)((td1 >> 16) | (((unsigned)tileCols & 0xFFFFu) << 16)); // tile_dim0
    g1[4] = (int)((unsigned)tileRows & 0xFFFFu);        // tile_dim1 (tile_dim2=0)
    g1[5] = (int)(st0 & 0xFFFFFFFFull);                 // dim0_stride[31:0]
    g1[6] = (int)((st0 >> 32) & 0xFFFFull);             // dim0_stride[47:32]
    g1[7] = 0;
    i32x4 z4 = {0, 0, 0, 0};                            // groups 2/3 unused (2D)
    i32x8 z8 = {0, 0, 0, 0, 0, 0, 0, 0};                // unused trailing group
    __builtin_amdgcn_tensor_load_to_lds(g0, g1, z4, z4, z8, 0);
}

// ---------------------------------------------------------------------------
__global__ void k_init(unsigned int* wsu) {
    if (threadIdx.x < 2) wsu[threadIdx.x] = 0u;         // absmax(x), absmax(y)
}

__global__ void k_absmax(const float4* __restrict__ x4, unsigned int* wsu, int n4) {
    __shared__ float sm[8];
    float m = 0.f;
    for (int i = blockIdx.x * blockDim.x + threadIdx.x; i < n4;
         i += gridDim.x * blockDim.x) {
        float4 v = x4[i];
        m = fmaxf(m, fmaxf(fmaxf(fabsf(v.x), fabsf(v.y)),
                           fmaxf(fabsf(v.z), fabsf(v.w))));
    }
    m = waveMax(m);
    int lane = threadIdx.x & 31, w = threadIdx.x >> 5;
    if (lane == 0) sm[w] = m;
    __syncthreads();
    if (threadIdx.x == 0) {
        float bm = sm[0];
#pragma unroll
        for (int i = 1; i < 8; ++i) bm = fmaxf(bm, sm[i]);
        atomicMax(wsu, __float_as_uint(bm));            // bits-compare ok: bm >= 0
    }
}

// One workgroup = 16 rows, 256 threads = 8 waves. Each wave TDM-loads its own
// 2-row slice of the 16x1024 chunk, then computes Ex/Ex2 via WMMA with an
// all-ones B (K-assignment is free -> lane-rotated columns, bank-conflict-free).
__global__ void k_rowstats(const float* __restrict__ x,
                           const float* __restrict__ gamma,
                           const float* __restrict__ beta,
                           unsigned int* wsu, float* wsMu, float* wsInv) {
    extern __shared__ float tile[];                     // 16*1024 f32 = 64 KB
    __shared__ float sumX[16], sumS[16], muS[16], invS[16], sm[8];

    const int tid  = (int)threadIdx.x;
    const int lane = tid & 31;
    const int wave = tid >> 5;
    const int mrow = lane & 15;                         // WMMA A: M = lane%16
    const int rowBase = (int)blockIdx.x * ROWS_PER_BLK;
    const int wv = __builtin_amdgcn_readfirstlane(wave); // force SGPR (per-wave)

    const float maxX       = __uint_as_float(wsu[0]);
    const float scaleIn    = maxX * (1.f / 127.f);
    const float invScaleIn = 1.f / scaleIn;

    if (tid < 16) { sumX[tid] = 0.f; sumS[tid] = 0.f; }

    v16h ones;
#pragma unroll
    for (int j = 0; j < 16; ++j) ones[j] = (_Float16)1.f;
    v8f cx = {};                                        // f32 accum: sum(x_int)
    v8f cs = {};                                        // f32 accum: sum(sq_ref)

    const unsigned ldsBase = (unsigned)(uintptr_t)(void*)tile;

    // ---- phase 1: Ex / Ex2 via WMMA over 4 chunks --------------------------
    for (int ch = 0; ch < N_COLS / CHUNK; ++ch) {
        tdmLoadTile(ldsBase + (unsigned)(wv * 2 * CHUNK * 4),
                    x + (size_t)(rowBase + wv * 2) * N_COLS + ch * CHUNK,
                    CHUNK, 2, N_COLS);
        __builtin_amdgcn_s_wait_tensorcnt(0);
        __syncthreads();

        const int S = wave * (CHUNK / 8);               // wave's 128-col K slice
#pragma unroll
        for (int it = 0; it < (CHUNK / 8) / 32; ++it) {
            const int kb = S + it * 32;
            v16h ax, as;
#pragma unroll
            for (int j = 0; j < 16; ++j) {
                int c = kb + ((lane + j) & 31);         // rotation: no bank conflicts,
                float xf = tile[mrow * CHUNK + c];      // valid since B == ones
                float xi = quantInt(xf, invScaleIn);    // x_int (exact in f16)
                ax[j] = (_Float16)xi;
                as[j] = (_Float16)sqRefOf(fabsf(xi) * scaleIn); // exact in f16
            }
            cx = __builtin_amdgcn_wmma_f32_16x16x32_f16(
                     false, ax, false, ones, (short)0, cx, false, false);
            cs = __builtin_amdgcn_wmma_f32_16x16x32_f16(
                     false, as, false, ones, (short)0, cs, false, false);
        }
        __syncthreads();                                // tile reused next chunk
    }

    // D layout: VGPR r, lanes 0-15 -> M=r ; lanes 16-31 -> M=r+8 (any column)
    if ((lane & 15) == 0) {
        int half = lane >> 4;
#pragma unroll
        for (int r = 0; r < 8; ++r) {
            atomicAdd(&sumX[r + 8 * half], cx[r]);      // ds_add_f32
            atomicAdd(&sumS[r + 8 * half], cs[r]);
        }
    }
    __syncthreads();

    // ---- per-row stats -----------------------------------------------------
    if (tid < 16) {
        float Ex  = scaleIn * sumX[tid];
        float Ex2 = sumS[tid];
        float mu  = Ex * (1.f / (float)N_COLS);
        float var = Ex2 * (1.f / (float)N_COLS) - mu * mu;
        float vi  = fminf(fmaxf(rintf(var), 1.f), 65535.f);
        int msb   = 31 - __clz((int)vi);                // exact floor(log2), <=15
        float inv = kInvSqrtLut[msb] * (1.f / 65536.f);
        muS[tid] = mu;  invS[tid] = inv;
        wsMu[rowBase + tid] = mu;  wsInv[rowBase + tid] = inv;
    }
    __syncthreads();

    // ---- phase 2: global max|y| (re-stage tiles, L2-hot) -------------------
    float lmax = 0.f;
    for (int ch = 0; ch < N_COLS / CHUNK; ++ch) {
        tdmLoadTile(ldsBase + (unsigned)(wv * 2 * CHUNK * 4),
                    x + (size_t)(rowBase + wv * 2) * N_COLS + ch * CHUNK,
                    CHUNK, 2, N_COLS);
        __builtin_amdgcn_s_wait_tensorcnt(0);
        __syncthreads();
        for (int i = tid; i < ROWS_PER_BLK * CHUNK; i += 256) {
            int m   = i >> 10;                          // CHUNK == 1024
            int c   = i & (CHUNK - 1);
            int col = ch * CHUNK + c;
            float xi = quantInt(tile[i], invScaleIn);
            float y  = fmaf((xi * scaleIn - muS[m]) * invS[m], gamma[col], beta[col]);
            lmax = fmaxf(lmax, fabsf(y));
        }
        __syncthreads();
    }
    lmax = waveMax(lmax);
    if (lane == 0) sm[wave] = lmax;
    __syncthreads();
    if (tid == 0) {
        float bm = sm[0];
#pragma unroll
        for (int i = 1; i < 8; ++i) bm = fmaxf(bm, sm[i]);
        atomicMax(&wsu[1], __float_as_uint(bm));
    }
}

__global__ void k_output(const float4* __restrict__ x4,
                         const float4* __restrict__ g4,
                         const float4* __restrict__ b4,
                         const unsigned int* __restrict__ wsu,
                         const float* __restrict__ wsMu,
                         const float* __restrict__ wsInv,
                         float4* __restrict__ out4) {
    const int row = (int)blockIdx.y;
    const int c4  = (int)blockIdx.x * (int)blockDim.x + (int)threadIdx.x; // 0..1023
    const float maxX        = __uint_as_float(wsu[0]);
    const float maxY        = __uint_as_float(wsu[1]);
    const float scaleIn     = maxX * (1.f / 127.f);
    const float invScaleIn  = 1.f / scaleIn;
    const float scaleOut    = maxY * (1.f / 127.f);
    const float invScaleOut = 1.f / scaleOut;
    const float mu  = wsMu[row];
    const float inv = wsInv[row];

    float4 xv = x4[(size_t)row * (N_COLS / 4) + c4];
    float4 gv = g4[c4];
    float4 bv = b4[c4];

    auto comp = [&](float xf, float g, float b) -> float {
        float xi = quantInt(xf, invScaleIn);
        float y  = fmaf((xi * scaleIn - mu) * inv, g, b);
        float yi = fminf(fmaxf(rintf(y * invScaleOut), -127.f), 127.f);
        return yi * scaleOut;
    };
    float4 r;
    r.x = comp(xv.x, gv.x, bv.x);
    r.y = comp(xv.y, gv.y, bv.y);
    r.z = comp(xv.z, gv.z, bv.z);
    r.w = comp(xv.w, gv.w, bv.w);
    out4[(size_t)row * (N_COLS / 4) + c4] = r;
}

// ---------------------------------------------------------------------------
extern "C" void kernel_launch(void* const* d_in, const int* in_sizes, int n_in,
                              void* d_out, int out_size, void* d_ws, size_t ws_size,
                              hipStream_t stream) {
    const float* x     = (const float*)d_in[0];
    const float* gamma = (const float*)d_in[1];
    const float* beta  = (const float*)d_in[2];
    float* out         = (float*)d_out;

    unsigned int* wsu = (unsigned int*)d_ws;
    float* wsf        = (float*)d_ws;
    float* wsMu       = wsf + 16;            // 8192 floats
    float* wsInv      = wsf + 16 + N_ROWS;   // 8192 floats

    k_init<<<1, 32, 0, stream>>>(wsu);

    const int n4 = N_ROWS * N_COLS / 4;
    k_absmax<<<4096, 256, 0, stream>>>((const float4*)x, wsu, n4);

    k_rowstats<<<N_ROWS / ROWS_PER_BLK, 256,
                 ROWS_PER_BLK * CHUNK * sizeof(float), stream>>>(
        x, gamma, beta, wsu, wsMu, wsInv);

    k_output<<<dim3(N_COLS / 4 / 256, N_ROWS), 256, 0, stream>>>(
        (const float4*)x, (const float4*)gamma, (const float4*)beta,
        wsu, wsMu, wsInv, (float4*)out);
}
